// NiNoModel_45586782880345
// MI455X (gfx1250) — matrix operand order repaired and checked by reading, not compiled, not run on hardware
//
#include <hip/hip_runtime.h>

// ---------------- model constants (fixed by the reference) ----------------
#define HID    128
#define LPE    8
#define MF     9
#define CTX    5
#define SEQ    40
#define OUTD   (MF * SEQ)   // 360
#define OUTDP  368          // 360 padded to 16
#define EAK    (MF * CTX)   // 45
#define NLAYER 3
#define TILE   16           // rows (edges/nodes) per wave-tile

typedef __attribute__((ext_vector_type(16))) __bf16 v16bf;
typedef __attribute__((ext_vector_type(8)))  __bf16 v8bf;
typedef __attribute__((ext_vector_type(4)))  __bf16 v4bf;
typedef __attribute__((ext_vector_type(8)))  float  v8f;

__device__ __forceinline__ float silu_f(float v) {
  return v * (1.0f / (1.0f + __expf(-v)));
}

#define WMMA_BF16(a, b, c) \
  __builtin_amdgcn_wmma_f32_16x16x32_bf16(false, (a), false, (b), (short)0, (c), false, false)

// ---------------------------------------------------------------------------
// A fragment (16x32 bf16, MxK) from LDS tile A[16][ldk] at K-offset kb.
// Contiguous K order (matches packed B): each lane reads 16 contiguous elements:
//   lanes 0-15 : K = kb+0..15   (M = lane)
//   lanes 16-31: K = kb+16..31  (M = lane-16)
__device__ __forceinline__ v16bf a_frag(const __bf16* A, int ldk, int lane, int kb) {
  const int m  = lane & 15;
  const int kg = (lane >> 4) << 4;  // 0 or 16
  const __bf16* row = A + m * ldk + kb + kg;
  v8bf lo = *(const v8bf*)(row);
  v8bf hi = *(const v8bf*)(row + 8);
  return __builtin_shufflevector(lo, hi, 0, 1, 2, 3, 4, 5, 6, 7,
                                 8, 9, 10, 11, 12, 13, 14, 15);
}

// B fragment from pre-packed bf16 weights P[kt][n][half][16]:
//   value(kt,n,half,i) = W[n][kt*32 + half*16 + i]  (zero-padded at pack time)
// One unconditional 32-byte contiguous load per lane.
__device__ __forceinline__ v16bf b_frag(const __bf16* P, int Npad, int lane, int kt, int n0) {
  const __bf16* p = P + ((((long)kt * Npad + n0 + (lane & 15)) << 1) + (lane >> 4)) * 16;
  return *(const v16bf*)p;
}

// float4 -> 4x bf16 packed LDS store (8 bytes)
__device__ __forceinline__ void cvt_store4(__bf16* dst, float4 v) {
  v4bf o;
  o[0] = (__bf16)v.x; o[1] = (__bf16)v.y; o[2] = (__bf16)v.z; o[3] = (__bf16)v.w;
  *(v4bf*)dst = o;
}

// ---------------------------------------------------------------------------
// Pack W (row-major [Ncols][K] fp32) into fragment-native bf16 layout with
// zero padding to [Kpad x Npad]. Linear out index = kt*Npad*32 + n*32 + h*16 + e.
__global__ void pack_w_kernel(const float* __restrict__ W, __bf16* __restrict__ P,
                              int Ncols, int K, int Npad, int Kpad) {
  long i = (long)blockIdx.x * blockDim.x + threadIdx.x;
  const long total = (long)Kpad * Npad;
  if (i >= total) return;
  const int  e    = (int)(i & 15);
  const int  h    = (int)((i >> 4) & 1);
  const long rest = i >> 5;
  const int  n    = (int)(rest % Npad);
  const int  kt   = (int)(rest / Npad);
  const int  k    = kt * 32 + h * 16 + e;
  const float v = (n < Ncols && k < K) ? W[(long)n * K + k] : 0.0f;
  P[i] = (__bf16)v;
}

__global__ void zero_kernel(float* p, long n) {
  long i      = (long)blockIdx.x * blockDim.x + threadIdx.x;
  long stride = (long)gridDim.x * blockDim.x;
  for (; i < n; i += stride) p[i] = 0.0f;
}

__global__ void deg_kernel(const int* __restrict__ dst, float* __restrict__ deg, int E) {
  int i = blockIdx.x * blockDim.x + threadIdx.x;
  if (i < E) atomicAdd(&deg[dst[i]], 1.0f);
}

// x = wte[pos_w] + pos @ node_W.T + node_b   (K=8 -> plain VALU, negligible)
__global__ void node_init_kernel(const float* __restrict__ pos, const int* __restrict__ pos_w,
                                 const float* __restrict__ wte, const float* __restrict__ nW,
                                 const float* __restrict__ nb, float* __restrict__ x, int N) {
  long i = (long)blockIdx.x * blockDim.x + threadIdx.x;
  if (i >= (long)N * HID) return;
  const int n = (int)(i / HID), c = (int)(i % HID);
  float acc = nb[c];
  const float* pr = pos + (long)n * LPE;
  const float* wr = nW + (long)c * LPE;
#pragma unroll
  for (int k = 0; k < LPE; ++k) acc += pr[k] * wr[k];
  x[i] = acc + wte[(long)pos_w[n] * HID + c];
}

// e = layer_embed[edge_type] + edge_attr @ edge_W.T + edge_b  (WMMA, K=45 pad 64)
__global__ void edge_init_kernel(const float* __restrict__ edge_attr, const int* __restrict__ edge_type,
                                 const float* __restrict__ layer_embed, const __bf16* __restrict__ pW,
                                 const float* __restrict__ eb, __bf16* __restrict__ ebuf, int E) {
  __shared__ __align__(16) __bf16 As[TILE][64];
  __shared__ __align__(16) __bf16 Os[TILE][HID];
  __shared__ int et[TILE];
  const int  lane = threadIdx.x;
  const long e0   = (long)blockIdx.x * TILE;

  if (lane < TILE) {
    long e = e0 + lane; if (e >= E) e = E - 1;
    et[lane] = edge_type[e];
  }
  // 16 rows x 45 floats are contiguous in memory for a 16-edge tile
  const float* base = edge_attr + e0 * EAK;
  for (int i = lane; i < TILE * EAK; i += 32)
    As[i / EAK][i % EAK] = (__bf16)base[i];
  for (int i = lane; i < TILE * (64 - EAK); i += 32)
    As[i / (64 - EAK)][EAK + i % (64 - EAK)] = (__bf16)0.0f;
  __syncthreads();

  for (int nt = 0; nt < 8; ++nt) {
    v8f acc = {0.f, 0.f, 0.f, 0.f, 0.f, 0.f, 0.f, 0.f};
#pragma unroll
    for (int kt = 0; kt < 2; ++kt) {
      v16bf a = a_frag(&As[0][0], 64, lane, kt * 32);
      v16bf b = b_frag(pW, HID, lane, kt, nt * 16);
      acc = WMMA_BF16(a, b, acc);
    }
    const int n = nt * 16 + (lane & 15);
#pragma unroll
    for (int v = 0; v < 8; ++v) {
      const int m = v + ((lane >> 4) << 3);
      Os[m][n] = (__bf16)(acc[v] + layer_embed[(long)et[m] * HID + n] + eb[n]);
    }
  }
  __syncthreads();
  for (int i = lane; i < TILE * (HID / 8); i += 32) {
    const int r = i >> 4, c = (i & 15) << 3;
    *(v8bf*)&ebuf[(e0 + r) * HID + c] = *(const v8bf*)&Os[r][c];
  }
}

// One PNA layer, per-edge part (fully fused):
//   e' = silu([x_s|x_d|e] @ We.T + be)        -> ebuf (bf16) + LDS A-tile
//   m  = silu([x_s|x_d|e'] @ Wpre.T + bpre)   -> atomic mean-agg numerator
__global__ void layer_edge_kernel(const float* __restrict__ x, __bf16* __restrict__ ebuf,
                                  const int* __restrict__ src, const int* __restrict__ dst,
                                  const __bf16* __restrict__ pWe, const float* __restrict__ be,
                                  const __bf16* __restrict__ pWpre, const float* __restrict__ bpre,
                                  float* __restrict__ agg, int E) {
  __shared__ __align__(16) __bf16 As[TILE][3 * HID];
  __shared__ int sidx[TILE], didx[TILE];
  const int  lane = threadIdx.x;
  const long e0   = (long)blockIdx.x * TILE;

  if (lane < TILE) {
    long e = e0 + lane; if (e >= E) e = E - 1;
    sidx[lane] = src[e];
    didx[lane] = dst[e];
  }
  // warm L2 (global_prefetch) for the packed weights
  __builtin_prefetch((const char*)pWe + (long)lane * 2048, 0, 3);
  __builtin_prefetch((const char*)pWpre + (long)lane * 2048, 0, 3);
  __syncthreads();

  // stage [x_s | x_d] via float4 loads (chunks 0..31 -> src, 32..63 -> dst)
  for (int i = lane; i < TILE * 64; i += 32) {
    const int r = i >> 6, cc = i & 63;
    const float4 f = (cc < 32)
        ? ((const float4*)(x + (long)sidx[r] * HID))[cc]
        : ((const float4*)(x + (long)didx[r] * HID))[cc - 32];
    cvt_store4(&As[r][cc << 2], f);
  }
  // stage e: straight bf16 vector copy
  for (int i = lane; i < TILE * (HID / 8); i += 32) {
    const int r = i >> 4, c = (i & 15) << 3;
    *(v8bf*)&As[r][2 * HID + c] = *(const v8bf*)&ebuf[(e0 + r) * HID + c];
  }
  __syncthreads();

  // ---- GEMM 1: edge update (hold all 8 N-tiles in registers) ----
  v8f acc[8];
#pragma unroll
  for (int nt = 0; nt < 8; ++nt) acc[nt] = (v8f){0.f, 0.f, 0.f, 0.f, 0.f, 0.f, 0.f, 0.f};
  for (int kt = 0; kt < 12; ++kt) {
    v16bf a = a_frag(&As[0][0], 3 * HID, lane, kt * 32);
#pragma unroll
    for (int nt = 0; nt < 8; ++nt) {
      v16bf b = b_frag(pWe, HID, lane, kt, nt * 16);
      acc[nt] = WMMA_BF16(a, b, acc[nt]);
    }
  }
  __syncthreads();  // all A reads done before the e-part is overwritten with e'
#pragma unroll
  for (int nt = 0; nt < 8; ++nt) {
    const int n = nt * 16 + (lane & 15);
#pragma unroll
    for (int v = 0; v < 8; ++v) {
      const int m = v + ((lane >> 4) << 3);
      As[m][2 * HID + n] = (__bf16)silu_f(acc[nt][v] + be[n]);
    }
  }
  __syncthreads();
  // write e' back to global (vectorized bf16)
  for (int i = lane; i < TILE * (HID / 8); i += 32) {
    const int r = i >> 4, c = (i & 15) << 3;
    *(v8bf*)&ebuf[(e0 + r) * HID + c] = *(const v8bf*)&As[r][2 * HID + c];
  }

  // ---- GEMM 2: message + fused atomic aggregation (m never hits HBM) ----
  for (int nt = 0; nt < 8; ++nt) {
    v8f am = {0.f, 0.f, 0.f, 0.f, 0.f, 0.f, 0.f, 0.f};
    for (int kt = 0; kt < 12; ++kt) {
      v16bf a = a_frag(&As[0][0], 3 * HID, lane, kt * 32);
      v16bf b = b_frag(pWpre, HID, lane, kt, nt * 16);
      am = WMMA_BF16(a, b, am);
    }
    const int n = nt * 16 + (lane & 15);
#pragma unroll
    for (int v = 0; v < 8; ++v) {
      const int m = v + ((lane >> 4) << 3);
      atomicAdd(&agg[(long)didx[m] * HID + n], silu_f(am[v] + bpre[n]));
    }
  }
}

// Per-node part: x = [x | agg/deg] @ Wpost.T + bpost  (silu except last layer)
__global__ void node_post_kernel(float* __restrict__ x, const float* __restrict__ agg,
                                 const float* __restrict__ deg, const __bf16* __restrict__ pW,
                                 const float* __restrict__ bpost, int N, int apply_silu) {
  __shared__ __align__(16) __bf16 As[TILE][2 * HID];
  __shared__ float dinv[TILE];
  const int  lane = threadIdx.x;
  const long n0   = (long)blockIdx.x * TILE;

  if (lane < TILE) {
    long node = n0 + lane; if (node >= N) node = N - 1;
    dinv[lane] = 1.0f / fmaxf(deg[node], 1.0f);
  }
  __syncthreads();
  for (int i = lane; i < TILE * 64; i += 32) {
    const int r = i >> 6, cc = i & 63;
    long node = n0 + r; if (node >= N) node = N - 1;
    float4 f;
    if (cc < 32) {
      f = ((const float4*)(x + node * HID))[cc];
    } else {
      f = ((const float4*)(agg + node * HID))[cc - 32];
      const float s = dinv[r];
      f.x *= s; f.y *= s; f.z *= s; f.w *= s;
    }
    cvt_store4(&As[r][cc << 2], f);
  }
  __syncthreads();

  for (int nt = 0; nt < 8; ++nt) {
    v8f acc = {0.f, 0.f, 0.f, 0.f, 0.f, 0.f, 0.f, 0.f};
#pragma unroll
    for (int kt = 0; kt < 8; ++kt) {
      v16bf a = a_frag(&As[0][0], 2 * HID, lane, kt * 32);
      v16bf b = b_frag(pW, HID, lane, kt, nt * 16);
      acc = WMMA_BF16(a, b, acc);
    }
    const int col = nt * 16 + (lane & 15);
#pragma unroll
    for (int v = 0; v < 8; ++v) {
      const int  m    = v + ((lane >> 4) << 3);
      const long node = n0 + m;
      if (node < N) {
        float val = acc[v] + bpost[col];
        if (apply_silu) val = silu_f(val);
        x[node * HID + col] = val;
      }
    }
  }
}

// Fused readout: h = silu(e @ W1.T + b1) (stays in LDS), out = h @ W2.T + b2 + res
__global__ void out_kernel(const __bf16* __restrict__ ebuf, const float* __restrict__ edge_attr,
                           const __bf16* __restrict__ pW1, const float* __restrict__ b1,
                           const __bf16* __restrict__ pW2, const float* __restrict__ b2,
                           float* __restrict__ out, int E) {
  __shared__ __align__(16) __bf16 As[TILE][HID];
  const int  lane = threadIdx.x;
  const long e0   = (long)blockIdx.x * TILE;

  for (int i = lane; i < TILE * (HID / 8); i += 32) {
    const int r = i >> 4, c = (i & 15) << 3;
    *(v8bf*)&As[r][c] = *(const v8bf*)&ebuf[(e0 + r) * HID + c];
  }
  __syncthreads();

  // GEMM 1: h (all 8 N-tiles in registers, then overwrite the A-tile with h)
  v8f acc[8];
#pragma unroll
  for (int nt = 0; nt < 8; ++nt) acc[nt] = (v8f){0.f, 0.f, 0.f, 0.f, 0.f, 0.f, 0.f, 0.f};
#pragma unroll
  for (int kt = 0; kt < 4; ++kt) {
    v16bf a = a_frag(&As[0][0], HID, lane, kt * 32);
#pragma unroll
    for (int nt = 0; nt < 8; ++nt) {
      v16bf b = b_frag(pW1, HID, lane, kt, nt * 16);
      acc[nt] = WMMA_BF16(a, b, acc[nt]);
    }
  }
  __syncthreads();
#pragma unroll
  for (int nt = 0; nt < 8; ++nt) {
    const int n = nt * 16 + (lane & 15);
#pragma unroll
    for (int v = 0; v < 8; ++v) {
      const int m = v + ((lane >> 4) << 3);
      As[m][n] = (__bf16)silu_f(acc[nt][v] + b1[n]);
    }
  }
  __syncthreads();

  // GEMM 2: 368 padded columns (zero-padded weights), store-guard at 360
  for (int nt = 0; nt < 23; ++nt) {
    v8f ao = {0.f, 0.f, 0.f, 0.f, 0.f, 0.f, 0.f, 0.f};
#pragma unroll
    for (int kt = 0; kt < 4; ++kt) {
      v16bf a = a_frag(&As[0][0], HID, lane, kt * 32);
      v16bf b = b_frag(pW2, OUTDP, lane, kt, nt * 16);
      ao = WMMA_BF16(a, b, ao);
    }
    const int col = nt * 16 + (lane & 15);
    if (col < OUTD) {
      const int mf = col / SEQ;
#pragma unroll
      for (int v = 0; v < 8; ++v) {
        const int  m = v + ((lane >> 4) << 3);
        const long e = e0 + m;
        out[e * OUTD + col] = ao[v] + b2[col] + edge_attr[e * EAK + mf * CTX + (CTX - 1)];
      }
    }
  }
}

// ---------------------------------------------------------------------------
extern "C" void kernel_launch(void* const* d_in, const int* in_sizes, int n_in,
                              void* d_out, int out_size, void* d_ws, size_t ws_size,
                              hipStream_t stream) {
  const float* edge_attr   = (const float*)d_in[0];
  const int*   edge_type   = (const int*)  d_in[1];
  const float* pos         = (const float*)d_in[2];
  const int*   pos_w       = (const int*)  d_in[3];
  const int*   edge_index  = (const int*)  d_in[4];
  const float* layer_embed = (const float*)d_in[5];
  const float* wte         = (const float*)d_in[6];
  const float* node_W      = (const float*)d_in[7];
  const float* node_b      = (const float*)d_in[8];
  const float* edge_W      = (const float*)d_in[9];
  const float* edge_b      = (const float*)d_in[10];
  const float* gWe         = (const float*)d_in[11];
  const float* gbe         = (const float*)d_in[12];
  const float* gWpre       = (const float*)d_in[13];
  const float* gbpre       = (const float*)d_in[14];
  const float* gWpost      = (const float*)d_in[15];
  const float* gbpost      = (const float*)d_in[16];
  const float* oW1         = (const float*)d_in[17];
  const float* ob1         = (const float*)d_in[18];
  const float* oW2         = (const float*)d_in[19];
  const float* ob2         = (const float*)d_in[20];

  const int E = in_sizes[1];   // edge_type count
  const int N = in_sizes[3];   // pos_w count
  const int* src = edge_index;
  const int* dst = edge_index + E;

  // outputs: x [N,HID] then out [E,OUTD], fp32, fully rewritten each call
  float* x   = (float*)d_out;
  float* out = (float*)d_out + (long)N * HID;

  // workspace layout
  char*  wsb = (char*)d_ws;
  size_t off = 0;
  __bf16* ebuf = (__bf16*)(wsb + off); off += (size_t)E * HID * 2;
  off = (off + 63) & ~(size_t)63;
  float* agg = (float*)(wsb + off); off += (size_t)N * HID * 4;
  float* deg = (float*)(wsb + off); off += (size_t)N * 4;
  off = (off + 63) & ~(size_t)63;
  __bf16* pEdgeW = (__bf16*)(wsb + off); off += (size_t)64 * HID * 2;
  __bf16* pWe    = (__bf16*)(wsb + off); off += (size_t)NLAYER * 3 * HID * HID * 2;
  __bf16* pWpre  = (__bf16*)(wsb + off); off += (size_t)NLAYER * 3 * HID * HID * 2;
  __bf16* pWpost = (__bf16*)(wsb + off); off += (size_t)NLAYER * 2 * HID * HID * 2;
  __bf16* pW1    = (__bf16*)(wsb + off); off += (size_t)HID * HID * 2;
  __bf16* pW2    = (__bf16*)(wsb + off); off += (size_t)HID * OUTDP * 2;

  auto pack = [&](const float* W, __bf16* P, int Ncols, int K, int Npad, int Kpad) {
    const long total = (long)Kpad * Npad;
    pack_w_kernel<<<(int)((total + 255) / 256), 256, 0, stream>>>(W, P, Ncols, K, Npad, Kpad);
  };

  // pre-pack all weights into fragment-native bf16 (tiny; L2-resident afterwards)
  pack(edge_W, pEdgeW, HID, EAK, HID, 64);
  for (int l = 0; l < NLAYER; ++l) {
    pack(gWe + (long)l * HID * 3 * HID, pWe + (long)l * 3 * HID * HID, HID, 3 * HID, HID, 3 * HID);
    pack(gWpre + (long)l * HID * 3 * HID, pWpre + (long)l * 3 * HID * HID, HID, 3 * HID, HID, 3 * HID);
    pack(gWpost + (long)l * HID * 2 * HID, pWpost + (long)l * 2 * HID * HID, HID, 2 * HID, HID, 2 * HID);
  }
  pack(oW1, pW1, HID, HID, HID, HID);
  pack(oW2, pW2, OUTD, HID, OUTDP, HID);

  const int etiles = (E + TILE - 1) / TILE;
  const int ntiles = (N + TILE - 1) / TILE;

  // degrees (once)
  zero_kernel<<<256, 256, 0, stream>>>(deg, (long)N);
  deg_kernel<<<(E + 255) / 256, 256, 0, stream>>>(dst, deg, E);

  // embeddings
  node_init_kernel<<<(int)(((long)N * HID + 255) / 256), 256, 0, stream>>>(
      pos, pos_w, wte, node_W, node_b, x, N);
  edge_init_kernel<<<etiles, 32, 0, stream>>>(
      edge_attr, edge_type, layer_embed, pEdgeW, edge_b, ebuf, E);

  // PNA layers
  for (int l = 0; l < NLAYER; ++l) {
    zero_kernel<<<1024, 256, 0, stream>>>(agg, (long)N * HID);
    layer_edge_kernel<<<etiles, 32, 0, stream>>>(
        x, ebuf, src, dst,
        pWe + (long)l * 3 * HID * HID, gbe + (long)l * HID,
        pWpre + (long)l * 3 * HID * HID, gbpre + (long)l * HID,
        agg, E);
    node_post_kernel<<<ntiles, 32, 0, stream>>>(
        x, agg, deg,
        pWpost + (long)l * 2 * HID * HID, gbpost + (long)l * HID,
        N, (l < NLAYER - 1) ? 1 : 0);
  }

  // fused readout MLP + residual
  out_kernel<<<etiles, 32, 0, stream>>>(ebuf, edge_attr, pW1, ob1, pW2, ob2, out, E);
}
